// TernaryLinear_31825707663462
// MI455X (gfx1250) — matrix operand order repaired and checked by reading, not compiled, not run on hardware
//
#include <hip/hip_runtime.h>
#include <hip/hip_bf16.h>

// ---------------------------------------------------------------------------
// TernaryLinear: y = x @ sign(w)^T ;  M=8192, K=4096, N=16384 (fp32 in/out)
//
// MI455X plan (memory-bound: ~0.94 GB fp32 traffic vs ~1.1 TFLOP):
//   Pass 1: x -> f16, w -> sign(w) f16 into d_ws (staged weight 134MB < 192MB L2)
//   Pass 2: WMMA GEMM, BK=64, double-buffered LDS filled by
//           GLOBAL_LOAD_ASYNC_TO_LDS_B128 (ASYNCcnt), 1 barrier per K-stage,
//           copy of tile i+1 overlapped with 16 WMMAs of tile i.
// ---------------------------------------------------------------------------

typedef __attribute__((ext_vector_type(16))) _Float16 v16h;
typedef __attribute__((ext_vector_type(8)))  float    v8f;

#define K_DIM 4096
#define N_DIM 16384
#define M_DIM 8192

#define BM 128
#define BN 128
#define BK 64
#define AST 72                       // padded LDS row stride (halves): conflict-free
#define TILE_HALVES (128 * AST)      // 9216 halves = 18432 B per matrix per buffer
#define TILE_BYTES  (TILE_HALVES * 2)
#define BUF_BYTES   (2 * TILE_BYTES) // A+B per buffer
#define NK (K_DIM / BK)              // 64 K-stages

union H16 { v16h h; uint4 u[2]; };
union H8  { uint4 u; _Float16 h[8]; };

// ---------------- Pass 1a: x fp32 -> f16 ----------------
__global__ __launch_bounds__(256) void cvt_x_f16(const float* __restrict__ x,
                                                 _Float16* __restrict__ xh, int n) {
  int i = (blockIdx.x * 256 + threadIdx.x) * 8;
  if (i >= n) return;
  const float4* p = (const float4*)(x + i);
  float4 a = p[0], b = p[1];
  H8 r;
  r.h[0] = (_Float16)a.x; r.h[1] = (_Float16)a.y;
  r.h[2] = (_Float16)a.z; r.h[3] = (_Float16)a.w;
  r.h[4] = (_Float16)b.x; r.h[5] = (_Float16)b.y;
  r.h[6] = (_Float16)b.z; r.h[7] = (_Float16)b.w;
  *(uint4*)(xh + i) = r.u;
}

// ---------------- Pass 1b: w fp32 -> sign(w) f16 ----------------
__device__ __forceinline__ _Float16 signh(float v) {
  return (_Float16)((v > 0.0f) ? 1.0f : ((v < 0.0f) ? -1.0f : 0.0f));
}

__global__ __launch_bounds__(256) void quant_w_f16(const float* __restrict__ w,
                                                   _Float16* __restrict__ wh, int n) {
  int i = (blockIdx.x * 256 + threadIdx.x) * 8;
  if (i >= n) return;
  const float4* p = (const float4*)(w + i);
  float4 a = p[0], b = p[1];
  H8 r;
  r.h[0] = signh(a.x); r.h[1] = signh(a.y);
  r.h[2] = signh(a.z); r.h[3] = signh(a.w);
  r.h[4] = signh(b.x); r.h[5] = signh(b.y);
  r.h[6] = signh(b.z); r.h[7] = signh(b.w);
  *(uint4*)(wh + i) = r.u;
}

// ---- CDNA5 async copy: global -> LDS, 16B per lane, tracked by ASYNCcnt ----
__device__ __forceinline__ void async_b128(unsigned int lds_addr, const void* gaddr) {
  asm volatile("global_load_async_to_lds_b128 %0, %1, off"
               :: "v"(lds_addr), "v"(gaddr)
               : "memory");
}
__device__ __forceinline__ void wait_async0() {
  asm volatile("s_wait_asynccnt 0x0" ::: "memory");
}

// ---------------- Pass 2: WMMA GEMM ----------------
// A = xh [M][K] f16, B = wh [N][K] f16 (K contiguous), C = out [M][N] f32.
// 256 threads = 8 waves (2x4), block tile 128x128, wave tile 64x32.
__global__ __launch_bounds__(256) void ternary_gemm_wmma(
    const _Float16* __restrict__ A,
    const _Float16* __restrict__ B,
    float* __restrict__ C) {
  // [buf0: A | B][buf1: A | B]
  __shared__ __align__(16) _Float16 Lds[4 * TILE_HALVES];

  const int tid  = threadIdx.x;
  const int lane = tid & 31;
  const int wave = tid >> 5;
  const int wm   = wave >> 2;     // 0..1
  const int wn   = wave & 3;      // 0..3

  const int bm0 = blockIdx.y * BM;
  const int bn0 = blockIdx.x * BN;

  // global->LDS mapping: tile = 128 rows x 64 halves = 1024 chunks of 16B,
  // 4 chunks per thread: c = tid + 256*q ; row = c>>3 ; col8 = (c&7)*8
  const _Float16* gA[4];
  const _Float16* gB[4];
  unsigned int ldsA[4], ldsB[4];
  const unsigned int ldsbase = (unsigned int)(size_t)(&Lds[0]);
#pragma unroll
  for (int q = 0; q < 4; ++q) {
    const int c    = tid + 256 * q;
    const int row  = c >> 3;
    const int col8 = (c & 7) * 8;
    gA[q]   = A + (size_t)(bm0 + row) * K_DIM + col8;
    gB[q]   = B + (size_t)(bn0 + row) * K_DIM + col8;
    ldsA[q] = ldsbase + (unsigned int)(row * AST + col8) * 2u;
    ldsB[q] = ldsA[q] + TILE_BYTES;
  }

  // prologue: async-copy tile 0 into buffer 0
#pragma unroll
  for (int q = 0; q < 4; ++q) {
    async_b128(ldsA[q], gA[q]);
    async_b128(ldsB[q], gB[q]);
    gA[q] += BK;
    gB[q] += BK;
  }

  v8f acc[4][2] = {};

  const int mwb = wm * 64;
  const int nwb = wn * 32;
  const int lr  = lane & 15;
  const int lhi = lane >> 4;

  for (int i = 0; i < NK; ++i) {
    wait_async0();        // this wave's tile-i chunks are in LDS
    __syncthreads();      // everyone's are

    // overlap: kick off tile i+1 into the other buffer
    if (i + 1 < NK) {
      const unsigned int bo = (unsigned int)(((i + 1) & 1) * BUF_BYTES);
#pragma unroll
      for (int q = 0; q < 4; ++q) {
        async_b128(ldsA[q] + bo, gA[q]);
        async_b128(ldsB[q] + bo, gB[q]);
        gA[q] += BK;
        gB[q] += BK;
      }
    }

    const _Float16* Abuf = Lds + (size_t)(i & 1) * 2 * TILE_HALVES;
    const _Float16* Bbuf = Abuf + TILE_HALVES;

#pragma unroll
    for (int s = 0; s < 2; ++s) {           // two 16x16x32 K-slices per stage
      // A frags: lanes 0-15: M=lr, K s*32+0..15 ; lanes 16-31: K s*32+16..31
      H16 af[4];
#pragma unroll
      for (int ii = 0; ii < 4; ++ii) {
        const _Float16* p = Abuf + (size_t)(mwb + ii * 16 + lr) * AST + s * 32 + lhi * 16;
        af[ii].u[0] = *(const uint4*)(p);
        af[ii].u[1] = *(const uint4*)(p + 8);
      }
      // B frags: lanes 0-15: N=lr, K s*32+0..15 ; lanes 16-31: K s*32+16..31
      H16 bf[2];
#pragma unroll
      for (int j = 0; j < 2; ++j) {
        const _Float16* p = Bbuf + (size_t)(nwb + j * 16 + lr) * AST + s * 32 + lhi * 16;
        bf[j].u[0] = *(const uint4*)(p);
        bf[j].u[1] = *(const uint4*)(p + 8);
      }
#pragma unroll
      for (int ii = 0; ii < 4; ++ii)
#pragma unroll
        for (int j = 0; j < 2; ++j)
          acc[ii][j] = __builtin_amdgcn_wmma_f32_16x16x32_f16(
              false, af[ii].h, false, bf[j].h,
              (short)0, acc[ii][j], false, false);
    }
  }

  // D layout: VGPR r -> lanes 0-15 (M=r, N=lane) ; lanes 16-31 (M=r+8, N=lane-16)
#pragma unroll
  for (int ii = 0; ii < 4; ++ii) {
    const int gm0 = bm0 + mwb + ii * 16 + lhi * 8;
#pragma unroll
    for (int j = 0; j < 2; ++j) {
      const int gn = bn0 + nwb + j * 16 + lr;
      float* o = C + (size_t)gm0 * N_DIM + gn;
#pragma unroll
      for (int r = 0; r < 8; ++r)
        o[(size_t)r * N_DIM] = acc[ii][j][r];
    }
  }
}

extern "C" void kernel_launch(void* const* d_in, const int* in_sizes, int n_in,
                              void* d_out, int out_size, void* d_ws, size_t ws_size,
                              hipStream_t stream) {
  const float* x = (const float*)d_in[0];   // [2,4096,4096] f32
  const float* w = (const float*)d_in[1];   // [16384,4096]  f32
  float* out = (float*)d_out;               // [2,4096,16384] f32

  _Float16* xh = (_Float16*)d_ws;                 // 64 MiB
  _Float16* wh = xh + (size_t)M_DIM * K_DIM;      // 128 MiB

  const int nx = M_DIM * K_DIM;
  const int nw = N_DIM * K_DIM;

  cvt_x_f16 <<<nx / (256 * 8), 256, 0, stream>>>(x, xh, nx);
  quant_w_f16<<<nw / (256 * 8), 256, 0, stream>>>(w, wh, nw);

  dim3 grid(N_DIM / BN, M_DIM / BM);              // 128 x 64 workgroups
  ternary_gemm_wmma<<<grid, 256, 0, stream>>>(xh, wh, out);
}
